// View_selector_RI_CAM_34961033789528
// MI455X (gfx1250) — matrix operand order repaired and checked by reading, not compiled, not run on hardware
//
#include <hip/hip_runtime.h>
#include <hip/hip_bf16.h>
#include <math.h>

// ---------------------------------------------------------------------------
// View_selector_RI_CAM for MI455X (gfx1250, wave32, WMMA)
//   B=4096, N=20, C=512, hidden=256, classes=40, S_VIEWS=8
// bf16 WMMA (16x16x32, f32 acc) for both MLP layers; weights staged as bf16
// transposed in LDS (292.8KB of the 320KB WGP LDS); hidden activations staged
// in LDS aliasing W1t (barrier-protected) for the layer-2 WMMA pass.
//
// Round 3 changes:
//  * FIXED fragment layouts per ISA tables:
//      A (16-bit 16x32): lane L: M=L%16; halfs[0..7]=K kb+8*(L/16)..+7,
//                        halfs[8..15]=K kb+16+8*(L/16)..+7  (second load +16h)
//      B (16-bit 32x16): lane L: N=L%16; halfs[0..7]=K kb+8*(L/16)..+7,
//                        halfs[8..15]=K kb+16+8*(L/16)..+7  (second load +16h)
//  * Pool kernel fused with f32->bf16 conversion of F0 (1 v_perm per pair);
//    big GEMM reads bf16 A directly -> no conversion VALU in the WMMA loop.
//    Selected at runtime only if ws_size is large enough (deterministic).
// ---------------------------------------------------------------------------

typedef __attribute__((ext_vector_type(16))) __bf16 v16bf;
typedef __attribute__((ext_vector_type(2)))  __bf16 v2bf;
typedef __attribute__((ext_vector_type(8)))  float  v8f;

struct alignas(16) U128 { unsigned int x[4]; };
struct alignas(16) F4   { float f[4]; };
struct alignas(8)  F2   { float f[2]; };

union BF16Frag { v16bf v; v2bf p[8]; unsigned short s[16]; U128 q[2]; };
union BF2      { v2bf v; unsigned short s[2]; unsigned int u; };

// Round-half-up f32->bf16 pair pack: one v_perm_b32 after two adds.
__device__ __forceinline__ unsigned int pack_bf16(float a, float b) {
  unsigned int ua = __float_as_uint(a) + 0x8000u;
  unsigned int ub = __float_as_uint(b) + 0x8000u;
  // D = {ub.b3, ub.b2, ua.b3, ua.b2} : low half = bf16(a), high half = bf16(b)
  return __builtin_amdgcn_perm(ub, ua, 0x07060302u);
}

__device__ __forceinline__ unsigned short f2bf(float f) {
  return (unsigned short)((__float_as_uint(f) + 0x8000u) >> 16);
}

#if __has_builtin(__builtin_amdgcn_cvt_pk_bf16_f32)
__device__ __forceinline__ v2bf f2bf_pk(float a, float b) {
  return __builtin_amdgcn_cvt_pk_bf16_f32(a, b);
}
#else
__device__ __forceinline__ v2bf f2bf_pk(float a, float b) {
  BF2 u; u.u = pack_bf16(a, b); return u.v;
}
#endif

__device__ __forceinline__ v8f wmma_bf16(v16bf a, v16bf b, v8f c) {
  // (neg_a, A, neg_b, B, c_mod, C, reuse_a, reuse_b)
  return __builtin_amdgcn_wmma_f32_16x16x32_bf16(false, a, false, b,
                                                 (short)0, c, false, false);
}

// ---- Fragment loaders (layouts per CDNA5 ISA 7.12.2) ----------------------
// A from global f32; caller passes p = &X[row*512 + kb + 8*lhi]
__device__ __forceinline__ v16bf load_a_f32(const float* __restrict__ p) {
  F4 q0 = *(const F4*)(p);
  F4 q1 = *(const F4*)(p + 4);
  F4 q2 = *(const F4*)(p + 16);
  F4 q3 = *(const F4*)(p + 20);
  BF16Frag u;
  u.p[0] = f2bf_pk(q0.f[0], q0.f[1]);
  u.p[1] = f2bf_pk(q0.f[2], q0.f[3]);
  u.p[2] = f2bf_pk(q1.f[0], q1.f[1]);
  u.p[3] = f2bf_pk(q1.f[2], q1.f[3]);
  u.p[4] = f2bf_pk(q2.f[0], q2.f[1]);
  u.p[5] = f2bf_pk(q2.f[2], q2.f[3]);
  u.p[6] = f2bf_pk(q3.f[0], q3.f[1]);
  u.p[7] = f2bf_pk(q3.f[2], q3.f[3]);
  return u.v;
}
// A from global/LDS bf16; p = base + row*stride + kb + 8*lhi (halfs)
__device__ __forceinline__ v16bf load_a_h(const unsigned short* p) {
  BF16Frag u;
  u.q[0] = *(const U128*)(p);        // K kb+8lhi   .. +7
  u.q[1] = *(const U128*)(p + 16);   // K kb+16+8lhi.. +7
  return u.v;
}
// B from LDS bf16; p = base + n*stride + kb + 8*lhi (halfs), same K order as A
__device__ __forceinline__ v16bf load_b_lds(const unsigned short* p) {
  BF16Frag u;
  u.q[0] = *(const U128*)(p);        // K kb+8lhi   .. +7
  u.q[1] = *(const U128*)(p + 16);   // K kb+16+8lhi.. +7
  return u.v;
}

// ---------------- LDS layout (dynamic shared, 292,800 bytes) ---------------
#define W1T_STRIDE 520                          // 512 + 8 pad halfs
#define W2T_STRIDE 264                          // 256 + 8 pad halfs
#define H_STRIDE   264
#define OFF_W1T 0
#define OFF_W2T (256 * W1T_STRIDE * 2)          // 266,240
#define OFF_B1  (OFF_W2T + 48 * W2T_STRIDE * 2) // 291,584
#define OFF_B2  (OFF_B1 + 256 * 4)              // 292,608
#define SMEM_TOTAL (OFF_B2 + 48 * 4)            // 292,800 <= 320KB WGP LDS

// 256 threads = 8 wave32; 128 rows x 256 hidden per block.
// Wave w: rowgrp = w&3 (2 M-tiles of 16), nhalf = w>>2 (8 N-tiles of 16).
// BF16A: A streamed as pre-converted bf16 (Xh); else f32 (Xf) + on-fly cvt.
template <bool BF16A>
__global__ void __launch_bounds__(256)
cls_gemm_wmma(const float* __restrict__ Xf, const unsigned short* __restrict__ Xh,
              const float* __restrict__ W1, const float* __restrict__ b1,
              const float* __restrict__ W2, const float* __restrict__ b2,
              float* __restrict__ out) {
  extern __shared__ char smem[];
  unsigned short* w1t  = (unsigned short*)(smem + OFF_W1T);
  unsigned short* w2t  = (unsigned short*)(smem + OFF_W2T);
  float*          b1s  = (float*)(smem + OFF_B1);
  float*          b2s  = (float*)(smem + OFF_B2);
  unsigned short* hbuf = (unsigned short*)(smem + OFF_W1T); // aliases w1t

  const int tid = threadIdx.x;

  // Stage W1^T bf16: w1t[n][k], k contiguous; packed k-pairs (4B LDS stores).
  for (int i = tid; i < 256 * 256; i += 256) {
    int kp = i >> 8, n = i & 255;            // kp = k/2
    float lo = W1[(2 * kp) * 256 + n];
    float hi = W1[(2 * kp + 1) * 256 + n];
    *(v2bf*)(w1t + n * W1T_STRIDE + 2 * kp) = f2bf_pk(lo, hi);
  }
  // Stage W2^T bf16 zero-padded to 48 output cols. W2 is [256][40].
  for (int i = tid; i < 48 * 128; i += 256) {
    int n = i >> 7, kp = i & 127;
    v2bf val;
    if (n < 40) {
      val = f2bf_pk(W2[(2 * kp) * 40 + n], W2[(2 * kp + 1) * 40 + n]);
    } else {
      BF2 z; z.u = 0; val = z.v;
    }
    *(v2bf*)(w2t + n * W2T_STRIDE + 2 * kp) = val;
  }
  b1s[tid] = b1[tid];
  if (tid < 40) b2s[tid] = b2[tid];
  __syncthreads();

  const int w    = tid >> 5;
  const int lane = tid & 31;
  const int lhi  = lane >> 4;     // K-half select
  const int lm   = lane & 15;     // M (A) or N (B/C/D) within tile
  const int rowgrp = w & 3;
  const int nhalf  = w >> 2;
  const int blockRow0 = blockIdx.x * 128;

  const v8f zero8 = {0.f, 0.f, 0.f, 0.f, 0.f, 0.f, 0.f, 0.f};
  v8f acc[2][8];
#pragma unroll
  for (int mt = 0; mt < 2; ++mt)
#pragma unroll
    for (int nt = 0; nt < 8; ++nt) acc[mt][nt] = zero8;

  const size_t row0 = (size_t)(blockRow0 + rowgrp * 32 + lm) * 512;
  const float*          xf0 = Xf + row0;
  const float*          xf1 = xf0 + 16 * 512;
  const unsigned short* xh0 = Xh + row0;
  const unsigned short* xh1 = xh0 + 16 * 512;

  // ---- Layer 1: [128,512] @ [512,256], K in 16 steps of 32 ----
  for (int kt = 0; kt < 16; ++kt) {
    const int kb = kt * 32;
    v16bf a0, a1;
    if constexpr (BF16A) {
      a0 = load_a_h(xh0 + kb + 8 * lhi);
      a1 = load_a_h(xh1 + kb + 8 * lhi);
    } else {
      a0 = load_a_f32(xf0 + kb + 8 * lhi);
      a1 = load_a_f32(xf1 + kb + 8 * lhi);
    }
#pragma unroll
    for (int nt = 0; nt < 8; ++nt) {
      int n = nhalf * 128 + nt * 16 + lm;
      v16bf bf = load_b_lds(w1t + n * W1T_STRIDE + kb + 8 * lhi);
      acc[0][nt] = wmma_bf16(a0, bf, acc[0][nt]);
      acc[1][nt] = wmma_bf16(a1, bf, acc[1][nt]);
    }
  }

  __syncthreads();   // all waves done reading w1t -> safe to alias with h

  // bias + LeakyReLU(0.2), write bf16 h[m][n] (m=0..127 block-local rows)
#pragma unroll
  for (int mt = 0; mt < 2; ++mt) {
#pragma unroll
    for (int nt = 0; nt < 8; ++nt) {
      int n = nhalf * 128 + nt * 16 + lm;
      float bias = b1s[n];
#pragma unroll
      for (int r = 0; r < 8; ++r) {
        float v = acc[mt][nt][r] + bias;
        v = v > 0.f ? v : 0.2f * v;
        int m = rowgrp * 32 + mt * 16 + r + 8 * lhi;  // C-layout row
        hbuf[m * H_STRIDE + n] = f2bf(v);
      }
    }
  }
  __syncthreads();

  // ---- Layer 2: [128,256] @ [256,48]; wave w owns M-tile w, 3 N-tiles ----
  v8f acc2[3];
#pragma unroll
  for (int nt = 0; nt < 3; ++nt) acc2[nt] = zero8;
  const unsigned short* ha = hbuf + (w * 16 + lm) * H_STRIDE;
  for (int kt = 0; kt < 8; ++kt) {
    const int kb = kt * 32;
    v16bf a = load_a_h(ha + kb + 8 * lhi);
#pragma unroll
    for (int nt = 0; nt < 3; ++nt) {
      v16bf bf = load_b_lds(w2t + (nt * 16 + lm) * W2T_STRIDE + kb + 8 * lhi);
      acc2[nt] = wmma_bf16(a, bf, acc2[nt]);
    }
  }
#pragma unroll
  for (int nt = 0; nt < 3; ++nt) {
    int n = nt * 16 + lm;
    if (n < 40) {
      float bias = b2s[n];
#pragma unroll
      for (int r = 0; r < 8; ++r) {
        int row = blockRow0 + w * 16 + r + 8 * lhi;
        out[(size_t)row * 40 + n] = acc2[nt][r] + bias;
      }
    }
  }
}

// pooled only (fallback when ws is too small for the bf16 copy of F0)
__global__ void pool_kernel(const float* __restrict__ F0, float* __restrict__ pooled) {
  int t = blockIdx.x * blockDim.x + threadIdx.x;   // < 4096*512
  int b = t >> 9, c = t & 511;
  const float* p = F0 + (size_t)b * 10240 + c;
  float s = 0.f;
#pragma unroll
  for (int n = 0; n < 20; ++n) s += p[n * 512];
  pooled[t] = s * 0.05f;
}

// pooled + bf16 copy of F0 (Xbf as packed u32 pairs), single pass over F0.
__global__ void pool_convert_kernel(const float* __restrict__ F0,
                                    float* __restrict__ pooled,
                                    unsigned int* __restrict__ Xbf) {
  int t = blockIdx.x * blockDim.x + threadIdx.x;   // < 4096*256
  int b = t >> 8, cp = t & 255;                    // column pair 2cp,2cp+1
  const float* p = F0 + (size_t)b * 10240 + 2 * cp;
  float s0 = 0.f, s1 = 0.f;
#pragma unroll
  for (int n = 0; n < 20; ++n) {
    F2 q = *(const F2*)(p + n * 512);
    s0 += q.f[0]; s1 += q.f[1];
    Xbf[((size_t)b * 20 + n) * 256 + cp] = pack_bf16(q.f[0], q.f[1]);
  }
  F2 r; r.f[0] = s0 * 0.05f; r.f[1] = s1 * 0.05f;
  *(F2*)(pooled + (size_t)b * 512 + 2 * cp) = r;
}

// pred[b] = argmax_j logits[b][j]  (first occurrence, like jnp.argmax)
__global__ void argmax_kernel(const float* __restrict__ logits, int* __restrict__ pred) {
  int b = blockIdx.x * blockDim.x + threadIdx.x;
  if (b >= 4096) return;
  const float* p = logits + (size_t)b * 40;
  float bv = p[0]; int bi = 0;
#pragma unroll
  for (int j = 1; j < 40; ++j) { float v = p[j]; if (v > bv) { bv = v; bi = j; } }
  pred[b] = bi;
}

// idx[b][0..7] = top-8 view indices of score[b,:,pred[b]] (descending, stable)
__global__ void topk_kernel(const float* __restrict__ score,
                            const int* __restrict__ pred, int* __restrict__ idx) {
  int b = blockIdx.x * blockDim.x + threadIdx.x;
  if (b >= 4096) return;
  int p = pred[b];
  float v[20];
#pragma unroll
  for (int n = 0; n < 20; ++n) v[n] = score[(size_t)b * 800 + n * 40 + p];
#pragma unroll
  for (int s = 0; s < 8; ++s) {
    float bv = v[0]; int bi = 0;
#pragma unroll
    for (int n = 1; n < 20; ++n) if (v[n] > bv) { bv = v[n]; bi = n; }
    idx[b * 8 + s] = bi;
    v[bi] = -INFINITY;
  }
}

// F_new[b][s][:] = F0[b][idx[b][s]][:]  (one float4 per thread)
__global__ void gather_f_kernel(const float* __restrict__ F0,
                                const int* __restrict__ idx, float* __restrict__ Fn) {
  int t = blockIdx.x * blockDim.x + threadIdx.x;   // < 32768*128
  int rowid = t >> 7, q = t & 127;
  int b = rowid >> 3, s = rowid & 7;
  int src = idx[b * 8 + s];
  const F4* in  = (const F4*)F0;
  F4*       out = (F4*)Fn;
  out[(size_t)rowid * 128 + q] = in[((size_t)b * 20 + src) * 128 + q];
}

// vertices_new[b][s][d] = vertices0[b][idx[b][s]][d]
__global__ void gather_v_kernel(const float* __restrict__ V0,
                                const int* __restrict__ idx, float* __restrict__ Vn) {
  int t = blockIdx.x * blockDim.x + threadIdx.x;   // < 98304
  if (t >= 4096 * 8 * 3) return;
  int rowid = t / 3, d = t - rowid * 3;
  int b = rowid >> 3, s = rowid & 7;
  int src = idx[b * 8 + s];
  Vn[t] = V0[((size_t)b * 20 + src) * 3 + d];
}

extern "C" void kernel_launch(void* const* d_in, const int* in_sizes, int n_in,
                              void* d_out, int out_size, void* d_ws, size_t ws_size,
                              hipStream_t stream) {
  (void)in_sizes; (void)n_in; (void)out_size;
  const float* F0 = (const float*)d_in[0];   // [4096,20,512]
  const float* V0 = (const float*)d_in[1];   // [4096,20,3]
  const float* W1 = (const float*)d_in[2];   // [512,256]
  const float* b1 = (const float*)d_in[3];   // [256]
  const float* W2 = (const float*)d_in[4];   // [256,40]
  const float* b2 = (const float*)d_in[5];   // [40]
  // d_in[6] = k (== S_VIEWS == 8), compile-time constant here

  float* out   = (float*)d_out;
  float* F_new = out;                                   // 4096*8*512
  float* score = out + (size_t)4096 * 8 * 512;          // 4096*20*40
  float* V_new = score + (size_t)4096 * 20 * 40;        // 4096*8*3

  char*  ws      = (char*)d_ws;
  size_t off_pooled = 0;                                   // 4096*512*4 = 8MB
  size_t off_logits = off_pooled + (size_t)4096 * 512 * 4;
  size_t off_pred   = off_logits + (size_t)4096 * 40 * 4;
  size_t off_idx    = off_pred + (size_t)4096 * 4;
  size_t off_xbf    = (off_idx + (size_t)4096 * 8 * 4 + 255) & ~(size_t)255;
  size_t need_bf16  = off_xbf + (size_t)81920 * 512 * 2;   // ~93 MB total

  float* pooled = (float*)(ws + off_pooled);
  float* logits = (float*)(ws + off_logits);
  int*   pred   = (int*)(ws + off_pred);
  int*   idx    = (int*)(ws + off_idx);
  unsigned int*   xbf = (unsigned int*)(ws + off_xbf);
  unsigned short* xh  = (unsigned short*)(ws + off_xbf);

  const bool use_bf16 = (ws_size >= need_bf16);   // fixed per process: deterministic

  // Allow >64KB dynamic LDS for the WMMA GEMM (292.8KB of 320KB WGP LDS).
  (void)hipFuncSetAttribute((const void*)&cls_gemm_wmma<false>,
                            hipFuncAttributeMaxDynamicSharedMemorySize, SMEM_TOTAL);
  (void)hipFuncSetAttribute((const void*)&cls_gemm_wmma<true>,
                            hipFuncAttributeMaxDynamicSharedMemorySize, SMEM_TOTAL);

  // 1) pooled_F = mean over views (+ bf16 copy of F0 when ws allows)
  if (use_bf16) {
    pool_convert_kernel<<<(4096 * 256) / 256, 256, 0, stream>>>(F0, pooled, xbf);
  } else {
    pool_kernel<<<(4096 * 512) / 256, 256, 0, stream>>>(F0, pooled);
  }
  // 2) logits = cls(pooled_F)   [4096 rows -> 32 blocks] (f32-A path, tiny)
  cls_gemm_wmma<false><<<4096 / 128, 256, SMEM_TOTAL, stream>>>(
      pooled, (const unsigned short*)nullptr, W1, b1, W2, b2, logits);
  // 3) score = cls(F0)          [81920 rows -> 640 blocks], straight to d_out
  if (use_bf16) {
    cls_gemm_wmma<true><<<81920 / 128, 256, SMEM_TOTAL, stream>>>(
        (const float*)nullptr, xh, W1, b1, W2, b2, score);
  } else {
    cls_gemm_wmma<false><<<81920 / 128, 256, SMEM_TOTAL, stream>>>(
        F0, (const unsigned short*)nullptr, W1, b1, W2, b2, score);
  }
  // 4) pred = argmax(logits)
  argmax_kernel<<<4096 / 256, 256, 0, stream>>>(logits, pred);
  // 5) idx = top8 of score[:, :, pred]
  topk_kernel<<<4096 / 256, 256, 0, stream>>>(score, pred, idx);
  // 6) gathers
  gather_f_kernel<<<(32768 * 128) / 256, 256, 0, stream>>>(F0, idx, F_new);
  gather_v_kernel<<<(98304 + 255) / 256, 256, 0, stream>>>(V0, idx, V_new);
}
